// MHA_Attention_83184926589084
// MI455X (gfx1250) — matrix operand order repaired and checked by reading, not compiled, not run on hardware
//
#include <hip/hip_runtime.h>
#include <hip/hip_bf16.h>
#include <cstdint>

typedef __attribute__((ext_vector_type(16))) __bf16 v16bf;
typedef __attribute__((ext_vector_type(8)))  float  v8f;
typedef __attribute__((ext_vector_type(4)))  unsigned v4u;
typedef __attribute__((ext_vector_type(8)))  int      v8i;
typedef __attribute__((ext_vector_type(4)))  int      v4i;

union FragBF { v16bf v; unsigned u[8]; };

static constexpr int Bn = 4;
static constexpr int Tn = 2048;
static constexpr int Cn = 1024;
static constexpr int Hn = 16;
static constexpr int Dn = 64;
static constexpr int Mn = Bn * Tn;       // 8192
static constexpr int Nqkv = 3 * Cn;      // 3072

static __device__ __forceinline__ unsigned short f2bf(float f) {
    union { float f; unsigned u; } v; v.f = f;
    unsigned r = v.u + 0x7FFFu + ((v.u >> 16) & 1u);   // round-to-nearest-even
    return (unsigned short)(r >> 16);
}

static __device__ __forceinline__ v8f vzero() {
    v8f z;
#pragma unroll
    for (int i = 0; i < 8; ++i) z[i] = 0.0f;
    return z;
}

static __device__ __forceinline__ v8f wmma_bf16(v16bf a, v16bf b, v8f c) {
    return __builtin_amdgcn_wmma_f32_16x16x32_bf16(false, a, false, b, (short)0, c, false, false);
}

// ---- Tensor Data Mover: stage a (up to 2D) bf16 tile into LDS ----------
// D# per CDNA5 ISA ch.8: group0 = {count, lds_addr, global_addr, type=2},
// group1 = {data_size=2B, tensor dims, tile dims, dim0 stride}, groups 2/3 = 0.
// tile_d1 == 0 -> 1D tile. Dims/strides in elements (bf16).
// This toolchain exposes the 6-arg builtin: (v4u, v8i, v4i, v4i, v8i, i32 cpol).
static __device__ __forceinline__ void tdm_load_bf16_tile(
    unsigned lds_off, const void* gptr,
    unsigned tensor_d0, unsigned tensor_d1,
    unsigned tile_d0, unsigned tile_d1,
    unsigned long long stride0) {
    const unsigned long long ga = (unsigned long long)gptr;
    v4u g0;
    g0[0] = 1u;                                        // count=1, user mode
    g0[1] = lds_off;                                   // LDS byte address
    g0[2] = (unsigned)ga;                              // global_addr[31:0]
    g0[3] = (unsigned)((ga >> 32) & 0x01FFFFFFull)     // global_addr[56:32]
            | 0x80000000u;                             // type=2 ("image")
    v8i g1;
    g1[0] = (int)(1u << 16);                           // data_size=1 -> 2 bytes
    g1[1] = (int)((tensor_d0 & 0xFFFFu) << 16);        // tensor_dim0[15:0]
    g1[2] = (int)((tensor_d0 >> 16) | ((tensor_d1 & 0xFFFFu) << 16));
    g1[3] = (int)((tensor_d1 >> 16) | (tile_d0 << 16));// tile_dim0
    g1[4] = (int)(tile_d1 & 0xFFFFu);                  // tile_dim1 (tile_dim2=0)
    g1[5] = (int)(stride0 & 0xFFFFFFFFull);            // tensor_dim0_stride lo
    g1[6] = (int)(stride0 >> 32);                      // stride hi (stride1=0)
    g1[7] = 0;
    v4i z4; z4[0] = 0; z4[1] = 0; z4[2] = 0; z4[3] = 0;
    v8i z8;
#pragma unroll
    for (int i = 0; i < 8; ++i) z8[i] = 0;
    __builtin_amdgcn_tensor_load_to_lds(g0, g1, z4, z4, z8, 0);
}

// ---------------- fp32 -> bf16 conversion ----------------
__global__ __launch_bounds__(256) void cvt_f32_bf16(const float* __restrict__ in,
                                                    unsigned short* __restrict__ out, int n) {
    for (int i = blockIdx.x * 256 + threadIdx.x; i < n; i += gridDim.x * 256)
        out[i] = f2bf(in[i]);
}

// ---------------- bf16 GEMM: out = A[M,K] @ B[K,N] + bias ----------------
// MODE 0: write fp32 to outF[M,N]
// MODE 1: scatter QKV (N = 3C) into q/k/v [B,H,T,D] bf16
template <int MODE>
__global__ __launch_bounds__(256) void gemm_bf16_kernel(
    const unsigned short* __restrict__ A, const unsigned short* __restrict__ Bm,
    const float* __restrict__ bias, float* __restrict__ outF,
    unsigned short* __restrict__ qd, unsigned short* __restrict__ kd,
    unsigned short* __restrict__ vd, int M, int N, int K) {
    __shared__ unsigned short As[128 * 32];   // [row][k]  (staged by TDM)
    __shared__ unsigned short Bs[128 * 32];   // [n][k]    (manual transpose)

    const int tid  = threadIdx.x;
    const int lane = tid & 31;
    const int wave = tid >> 5;
    const int wm = wave >> 1, wn = wave & 1;
    const int lo16 = lane & 15;
    const int hi8  = (lane >> 4) * 8;
    const int kBaseA = (lane < 16) ? 0 : 8;
    const int kBaseB = (lane < 16) ? 0 : 16;
    const int m0 = blockIdx.y * 128;
    const int n0 = blockIdx.x * 128;

    v8f acc[2][4];
#pragma unroll
    for (int i = 0; i < 2; ++i)
#pragma unroll
        for (int j = 0; j < 4; ++j) acc[i][j] = vzero();

    const int bK = tid >> 3, bSeg = (tid & 7) * 16;      // B: 16 contig n/thread, scatter

    const unsigned asOff = (unsigned)(uintptr_t)As;

    for (int kt = 0; kt < K; kt += 32) {
        // A tile via Tensor Data Mover: 2D tile 32(k) x 128(rows), row stride K
        if (wave == 0)
            tdm_load_bf16_tile(asOff, A + (size_t)m0 * K + kt,
                               /*tensor_d0=*/32, /*tensor_d1=*/128,
                               /*tile_d0=*/32, /*tile_d1=*/128,
                               /*stride0=*/(unsigned long long)K);

        // B tile: manual transpose into [n][k]
        const unsigned short* bG = Bm + (size_t)(kt + bK) * N + n0 + bSeg;
#pragma unroll
        for (int j = 0; j < 16; ++j) Bs[(bSeg + j) * 32 + bK] = bG[j];

        __builtin_amdgcn_s_wait_tensorcnt(0);
        __syncthreads();

        FragBF af[2], bf[4];
#pragma unroll
        for (int mt = 0; mt < 2; ++mt) {
            const int row = wm * 32 + mt * 16 + lo16;
#pragma unroll
            for (int p = 0; p < 8; ++p) {
                const int k0 = kBaseB + 2 * p;
                af[mt].u[p] = *(const unsigned*)(As + row * 32 + k0);
            }
        }
#pragma unroll
        for (int nt = 0; nt < 4; ++nt) {
            const int ncol = wn * 64 + nt * 16 + lo16;
#pragma unroll
            for (int p = 0; p < 8; ++p)
                bf[nt].u[p] = *(const unsigned*)(Bs + ncol * 32 + kBaseB + 2 * p);
        }
#pragma unroll
        for (int mt = 0; mt < 2; ++mt)
#pragma unroll
            for (int nt = 0; nt < 4; ++nt)
                acc[mt][nt] = wmma_bf16(af[mt].v, bf[nt].v, acc[mt][nt]);

        __syncthreads();
    }

#pragma unroll
    for (int mt = 0; mt < 2; ++mt)
#pragma unroll
        for (int nt = 0; nt < 4; ++nt)
#pragma unroll
            for (int r = 0; r < 8; ++r) {
                const int m = m0 + wm * 32 + mt * 16 + r + hi8;
                const int n = n0 + wn * 64 + nt * 16 + lo16;
                const float val = acc[mt][nt][r] + bias[n];
                if (MODE == 0) {
                    outF[(size_t)m * N + n] = val;
                } else {
                    const int which = n >> 10;       // /C
                    const int c = n & (Cn - 1);
                    const int h = c >> 6, d = c & (Dn - 1);
                    const int b = m >> 11;           // /T
                    const int t = m & (Tn - 1);
                    unsigned short* dst = (which == 0) ? qd : (which == 1) ? kd : vd;
                    dst[((size_t)(b * Hn + h) * Tn + t) * Dn + d] = f2bf(val);
                }
            }
}

// ---------------- causal flash attention, bf16 WMMA ----------------
// grid: (T/64, B*H); block: 128 (4 waves, 16 q-rows each)
__global__ __launch_bounds__(128) void attn_kernel(const unsigned short* __restrict__ q,
                                                   const unsigned short* __restrict__ k,
                                                   const unsigned short* __restrict__ v,
                                                   unsigned short* __restrict__ o) {
    __shared__ unsigned short Ks[64 * 64];   // [key][d]  (staged by TDM, contiguous)
    __shared__ unsigned short Vt[64 * 64];   // [d][key]  (manual transpose)
    __shared__ unsigned short Ps[64 * 64];   // [qrow][key], per-wave 16-row slices

    const int tid = threadIdx.x;
    const int lane = tid & 31;
    const int wave = tid >> 5;
    const int lo16 = lane & 15;
    const int hi8 = (lane >> 4) * 8;
    const int kBaseA = (lane < 16) ? 0 : 8;
    const int kBaseB = (lane < 16) ? 0 : 16;

    const int bh = blockIdx.y;
    const int b = bh >> 4, h = bh & (Hn - 1);
    const int q0 = blockIdx.x * 64;

    const unsigned short* qp = q + (size_t)bh * Tn * Dn;
    const unsigned short* kp = k + (size_t)bh * Tn * Dn;
    const unsigned short* vp = v + (size_t)bh * Tn * Dn;

    // Q fragments (held for whole kernel)
    FragBF qf[2];
    {
        const int qr = q0 + wave * 16 + lo16;
#pragma unroll
        for (int kt = 0; kt < 2; ++kt)
#pragma unroll
            for (int p = 0; p < 8; ++p) {
                const int d0 = kt * 32 + kBaseB + 2 * p;
                qf[kt].u[p] = *(const unsigned*)(qp + (size_t)qr * Dn + d0);
            }
    }

    float rmax[8], rsum[8];
    v8f oacc[4];
#pragma unroll
    for (int r = 0; r < 8; ++r) { rmax[r] = -3.0e38f; rsum[r] = 0.0f; }
#pragma unroll
    for (int dt = 0; dt < 4; ++dt) oacc[dt] = vzero();

    const unsigned ksOff = (unsigned)(uintptr_t)Ks;

    const int nKb = (q0 >> 6) + 1;   // causal: key blocks 0 .. q-block inclusive
    for (int kb = 0; kb < nKb; ++kb) {
        // K block [64 keys][64 d]: 8 KB contiguous -> 1D TDM load
        if (wave == 0)
            tdm_load_bf16_tile(ksOff, kp + (size_t)kb * 64 * Dn,
                               /*tensor_d0=*/4096, /*tensor_d1=*/0,
                               /*tile_d0=*/4096, /*tile_d1=*/0,
                               /*stride0=*/4096ull);
        {   // stage V block transposed: Vt[d][key]
            const int key = tid >> 1, dseg = (tid & 1) * 32;
            const unsigned short* src = vp + (size_t)(kb * 64 + key) * Dn + dseg;
#pragma unroll
            for (int j = 0; j < 32; ++j) Vt[(dseg + j) * 64 + key] = src[j];
        }
        __builtin_amdgcn_s_wait_tensorcnt(0);
        __syncthreads();

        // S = Q @ K^T  (16 q-rows x 64 keys per wave)
        v8f s[4];
#pragma unroll
        for (int nt = 0; nt < 4; ++nt) {
            s[nt] = vzero();
#pragma unroll
            for (int kt = 0; kt < 2; ++kt) {
                FragBF kf;
                const int key = nt * 16 + lo16;
#pragma unroll
                for (int p = 0; p < 8; ++p)
                    kf.u[p] = *(const unsigned*)(Ks + key * 64 + kt * 32 + kBaseB + 2 * p);
                s[nt] = wmma_bf16(qf[kt].v, kf.v, s[nt]);
            }
        }

        // scale + causal mask
#pragma unroll
        for (int nt = 0; nt < 4; ++nt) {
            const int kg = kb * 64 + nt * 16 + lo16;
#pragma unroll
            for (int r = 0; r < 8; ++r) {
                float val = s[nt][r] * 0.125f;   // 1/sqrt(64)
                const int qg = q0 + wave * 16 + r + hi8;
                if (kg > qg) val = -3.0e38f;
                s[nt][r] = val;
            }
        }

        // online softmax per row; write P (bf16) to LDS
#pragma unroll
        for (int r = 0; r < 8; ++r) {
            float mrow = s[0][r];
#pragma unroll
            for (int nt = 1; nt < 4; ++nt) mrow = fmaxf(mrow, s[nt][r]);
#pragma unroll
            for (int off = 1; off < 16; off <<= 1)
                mrow = fmaxf(mrow, __shfl_xor(mrow, off, 32));
            const float mnew = fmaxf(rmax[r], mrow);
            const float corr = __expf(rmax[r] - mnew);
            float lsum = 0.0f;
            const int prow = (wave * 16 + r + hi8) * 64;
#pragma unroll
            for (int nt = 0; nt < 4; ++nt) {
                const float e = __expf(s[nt][r] - mnew);
                lsum += e;
                Ps[prow + nt * 16 + lo16] = f2bf(e);
            }
#pragma unroll
            for (int off = 1; off < 16; off <<= 1)
                lsum += __shfl_xor(lsum, off, 32);
            rsum[r] = rsum[r] * corr + lsum;
            rmax[r] = mnew;
#pragma unroll
            for (int dt = 0; dt < 4; ++dt) oacc[dt][r] *= corr;
        }

        // O += P @ V
#pragma unroll
        for (int kt = 0; kt < 2; ++kt) {
            FragBF pf;
            const int prow = (wave * 16 + lo16) * 64;
#pragma unroll
            for (int p = 0; p < 8; ++p) {
                const int key = kt * 32 + kBaseB + 2 * p;
                pf.u[p] = *(const unsigned*)(Ps + prow + key);
            }
#pragma unroll
            for (int dt = 0; dt < 4; ++dt) {
                FragBF vf;
                const int d = dt * 16 + lo16;
#pragma unroll
                for (int p = 0; p < 8; ++p)
                    vf.u[p] = *(const unsigned*)(Vt + d * 64 + kt * 32 + kBaseB + 2 * p);
                oacc[dt] = wmma_bf16(pf.v, vf.v, oacc[dt]);
            }
        }
        __syncthreads();
    }

    // normalize + write attn output as bf16 [B*T, C]
#pragma unroll
    for (int dt = 0; dt < 4; ++dt)
#pragma unroll
        for (int r = 0; r < 8; ++r) {
            const float val = oacc[dt][r] / rsum[r];
            const int row = b * Tn + q0 + wave * 16 + r + hi8;
            const int col = h * Dn + dt * 16 + lo16;
            o[(size_t)row * Cn + col] = f2bf(val);
        }
}

// ---------------- launcher ----------------
extern "C" void kernel_launch(void* const* d_in, const int* in_sizes, int n_in,
                              void* d_out, int out_size, void* d_ws, size_t ws_size,
                              hipStream_t stream) {
    const float* x      = (const float*)d_in[0];
    const float* W_qkv  = (const float*)d_in[1];
    const float* b_qkv  = (const float*)d_in[2];
    const float* W_proj = (const float*)d_in[3];
    const float* b_proj = (const float*)d_in[4];

    unsigned short* ws16   = (unsigned short*)d_ws;
    const size_t nX = (size_t)Mn * Cn;            // 8388608
    unsigned short* xb     = ws16;                // x bf16; later reused as attn-out
    unsigned short* wqkvb  = xb + nX;
    unsigned short* wprojb = wqkvb + (size_t)Cn * Nqkv;
    unsigned short* qws    = wprojb + (size_t)Cn * Cn;
    unsigned short* kws    = qws + nX;
    unsigned short* vws    = kws + nX;

    cvt_f32_bf16<<<4096, 256, 0, stream>>>(x, xb, (int)nX);
    cvt_f32_bf16<<<4096, 256, 0, stream>>>(W_qkv, wqkvb, Cn * Nqkv);
    cvt_f32_bf16<<<2048, 256, 0, stream>>>(W_proj, wprojb, Cn * Cn);

    gemm_bf16_kernel<1><<<dim3(Nqkv / 128, Mn / 128), 256, 0, stream>>>(
        xb, wqkvb, b_qkv, nullptr, qws, kws, vws, Mn, Nqkv, Cn);

    attn_kernel<<<dim3(Tn / 64, Bn * Hn), 128, 0, stream>>>(qws, kws, vws, xb);

    gemm_bf16_kernel<0><<<dim3(Cn / 128, Mn / 128), 256, 0, stream>>>(
        xb, wprojb, b_proj, (float*)d_out, nullptr, nullptr, nullptr, Mn, Cn, Cn);
}
